// BiDirectionalCrossAttentionLayer_74766790689039
// MI455X (gfx1250) — matrix operand order repaired and checked by reading, not compiled, not run on hardware
//
#include <hip/hip_runtime.h>

typedef __attribute__((ext_vector_type(16))) __bf16 v16bf;
typedef __attribute__((ext_vector_type(8)))  __bf16 v8bf;
typedef __attribute__((ext_vector_type(8)))  float  v8f;
typedef __attribute__((ext_vector_type(4)))  unsigned int u32x4;
typedef __attribute__((ext_vector_type(8)))  int    i32x8;
typedef __attribute__((ext_vector_type(4)))  int    i32x4;

#define NSTR 4
#define BB   4
#define SS   512
#define EE   512
#define HH   8
#define HD   64

#if __has_include(<hip/amd_detail/amd_gfx1250_TDM.h>)
#define TDM_6ARG 1
#endif

// ---------------------------------------------------------------------------
// TDM: async DMA of a 2D bf16 tile (tile_d1 rows x tile_d0 elems, row stride
// in elements) from global into LDS at byte offset lds_off.
// D# bitfields per CDNA5 ISA ch.8.
__device__ __forceinline__ void tdm_load_2d_bf16(
    const __bf16* gptr, unsigned lds_off,
    unsigned tile_d0, unsigned tile_d1, unsigned row_stride_elems)
{
    unsigned long long ga = (unsigned long long)(size_t)gptr;
    u32x4 g0;
    g0[0] = 1u;                                   // count=1 (user descriptor)
    g0[1] = lds_off;                              // lds_addr (bytes)
    g0[2] = (unsigned)(ga & 0xFFFFFFFFu);         // global_addr[31:0]
    g0[3] = (unsigned)((ga >> 32) & 0x01FFFFFFu) | (2u << 30);  // addr hi | type=2
    i32x8 g1;
    g1[0] = (int)(1u << 16);                      // data_size=1 (2 bytes)
    g1[1] = (int)(tile_d0 << 16);                 // tensor_dim0[15:0]
    g1[2] = (int)((tile_d0 >> 16) | (tile_d1 << 16)); // td0 hi | tensor_dim1 lo
    g1[3] = (int)((tile_d1 >> 16) | (tile_d0 << 16)); // td1 hi | tile_dim0
    g1[4] = (int)(tile_d1 & 0xFFFFu);             // tile_dim1 (tile_dim2=0)
    g1[5] = (int)row_stride_elems;                // tensor_dim0_stride[31:0]
    g1[6] = 0;
    g1[7] = 0;
    i32x4 gz = {0, 0, 0, 0};
#if defined(TDM_6ARG)
    i32x8 gz8 = {0, 0, 0, 0, 0, 0, 0, 0};
    __builtin_amdgcn_tensor_load_to_lds(g0, g1, gz, gz, gz8, 0);
#else
    __builtin_amdgcn_tensor_load_to_lds(g0, g1, gz, gz, 0);
#endif
}

__device__ __forceinline__ unsigned lds_off_of(const void* p) {
    return (unsigned)(size_t)p;                   // flat LDS addr[31:0] = LDS offset
}

// 16-bit A 16x32 fragment in this code's own k order (B matches), lane row m = L&15,
// element e -> k = 16*half + e (one contiguous 16-element run).
__device__ __forceinline__ int a_frag_k(int e, int half) {
    int rr = e >> 1, p = e & 1;
    return 16 * half + 2 * rr + p;
}

// ---------------------------------------------------------------------------
__global__ __launch_bounds__(256) void stack_cvt(
    const float* __restrict__ x0, const float* __restrict__ x1,
    const float* __restrict__ x2, const float* __restrict__ x3,
    float* __restrict__ xs, __bf16* __restrict__ xb, int perStream)
{
    int ns  = blockIdx.y;
    int idx = blockIdx.x * 256 + threadIdx.x;
    if (idx >= perStream) return;
    const float* p = (ns == 0) ? x0 : (ns == 1) ? x1 : (ns == 2) ? x2 : x3;
    float v = p[idx];
    size_t o = (size_t)ns * perStream + idx;
    xs[o] = v;
    xb[o] = (__bf16)v;
}

__global__ __launch_bounds__(256) void cvt_bf16(
    const float* __restrict__ in, __bf16* __restrict__ out, long n)
{
    long idx = (long)blockIdx.x * 256 + threadIdx.x;
    if (idx < n) out[idx] = (__bf16)in[idx];
}

// ---------------------------------------------------------------------------
// Per-stream GEMM: out[ns] = A[ns](MxK) * W[ns](KxN) (+bias)(+GELU)
// 128 threads = 4 waves; block tile 64x64; K-step 32.
// A tile: TDM DMA (row-major). W tile: vectorized global_load_b128 into regs,
// then transposed ds_store_b16 scatter -> Bst[n][k] (contiguous B-fragments).
__global__ __launch_bounds__(128) void gemm_bf16(
    const __bf16* __restrict__ A, const __bf16* __restrict__ W,
    const float* __restrict__ bias, float* __restrict__ outF,
    __bf16* __restrict__ outB, int M, int N, int K, int actGelu)
{
    int ns = blockIdx.z;
    const __bf16* Ab = A + (size_t)ns * M * K;
    const __bf16* Wb = W + (size_t)ns * K * N;
    const float*  bb = bias ? bias + (size_t)ns * N : nullptr;
    int n0 = blockIdx.x * 64, m0 = blockIdx.y * 64;

    __shared__ __bf16 As[64 * 32];     // [m][k] row-major (TDM target)
    __shared__ __bf16 Bst[64 * 32];    // [n][k] transposed weight tile

    int t = threadIdx.x, wave = t >> 5, lane = t & 31;
    int half = lane >> 4, lq = lane & 15;
    // chunk ids for the transposed W load: 256 chunks of 8 elems, 2 per thread
    int kk0 = t >> 3,          nb0 = t & 7;
    int kk1 = (t + 128) >> 3,  nb1 = (t + 128) & 7;

    v8f c[4] = {};
    for (int k0 = 0; k0 < K; k0 += 32) {
        __syncthreads();
        if (wave == 0)
            tdm_load_2d_bf16(Ab + (size_t)m0 * K + k0, lds_off_of(As), 32, 64, K);
        // W tile 32x64: two 16B vector loads, then transposed scatter to LDS
        v8bf w0 = *(const v8bf*)&Wb[(size_t)(k0 + kk0) * N + n0 + nb0 * 8];
        v8bf w1 = *(const v8bf*)&Wb[(size_t)(k0 + kk1) * N + n0 + nb1 * 8];
        #pragma unroll
        for (int e = 0; e < 8; ++e) Bst[(nb0 * 8 + e) * 32 + kk0] = w0[e];
        #pragma unroll
        for (int e = 0; e < 8; ++e) Bst[(nb1 * 8 + e) * 32 + kk1] = w1[e];
        if (wave == 0) __builtin_amdgcn_s_wait_tensorcnt(0);
        __syncthreads();

        v16bf af;
        int mrow = wave * 16 + lq;
        #pragma unroll
        for (int e = 0; e < 16; ++e) af[e] = As[mrow * 32 + a_frag_k(e, half)];

        #pragma unroll
        for (int nt = 0; nt < 4; ++nt) {
            v16bf bf;
            int nc = nt * 16 + lq;
            #pragma unroll
            for (int e = 0; e < 16; ++e) bf[e] = Bst[nc * 32 + 16 * half + e];
            c[nt] = __builtin_amdgcn_wmma_f32_16x16x32_bf16(
                false, af, false, bf, (short)0, c[nt], false, false);
        }
    }

    #pragma unroll
    for (int nt = 0; nt < 4; ++nt) {
        int nc = n0 + nt * 16 + lq;
        float bv = bb ? bb[nc] : 0.f;
        #pragma unroll
        for (int r = 0; r < 8; ++r) {
            int mrow = m0 + wave * 16 + r + 8 * half;
            float v = c[nt][r] + bv;
            if (actGelu) v = 0.5f * v * (1.f + erff(v * 0.70710678118654752f));
            size_t oidx = (size_t)ns * M * N + (size_t)mrow * N + nc;
            if (outF) outF[oidx] = v;
            if (outB) outB[oidx] = (__bf16)v;
        }
    }
}

// ---------------------------------------------------------------------------
// Fused cross-stream flash attention.
// grid: (S/64, B*H, NS-i). block: 128 = 4 waves, 16 q-rows/wave.
// K tile: TDM (row-major). V tile: vector global loads + transposed LDS store.
__global__ __launch_bounds__(128) void attn_kernel(
    const __bf16* __restrict__ Q, const __bf16* __restrict__ Kb,
    const __bf16* __restrict__ Vb, const float* __restrict__ inter,
    __bf16* __restrict__ mh2)
{
    int i  = blockIdx.z;
    int b  = blockIdx.y / HH, h = blockIdx.y % HH;
    int q0 = blockIdx.x * 64;
    int t = threadIdx.x, wave = t >> 5, lane = t & 31;
    int half = lane >> 4, lq = lane & 15;

    __shared__ __bf16 Kt[64 * 64];     // [key][d]  (TDM target)
    __shared__ __bf16 Vtt[64 * 64];    // [d][key]  transposed
    __shared__ __bf16 Pt[4 * 16 * 64]; // per-wave P tile

    v16bf aq[2];
    size_t qbase = (((size_t)i * BB + b) * SS) * EE + (size_t)h * HD;
    {
        int mrow = q0 + wave * 16 + lq;
        #pragma unroll
        for (int kt = 0; kt < 2; ++kt)
            #pragma unroll
            for (int e = 0; e < 16; ++e)
                aq[kt][e] = Q[qbase + (size_t)mrow * EE + a_frag_k(e, half) + 32 * kt];
    }

    v8f o[4] = {};
    for (int j = 0; j < NSTR; ++j) {
        float wij = inter[i * NSTR + j] * 0.125f;   // SCALE = 64^-0.5
        size_t kvbase = (((size_t)j * BB + b) * SS) * EE + (size_t)h * HD;
        v8f oj[4] = {};
        float mrun[8], lrun[8];
        #pragma unroll
        for (int r = 0; r < 8; ++r) { mrun[r] = -1e30f; lrun[r] = 0.f; }

        for (int kb = 0; kb < SS / 64; ++kb) {
            __syncthreads();
            if (wave == 0)
                tdm_load_2d_bf16(Kb + kvbase + (size_t)kb * 64 * EE,
                                 lds_off_of(Kt), 64, 64, EE);
            // V tile 64x64: 4 chunks/thread of 8 contiguous bf16, then
            // transposed scatter into Vtt[d][key]
            {
                v8bf tv[4];
                #pragma unroll
                for (int cc = 0; cc < 4; ++cc) {
                    int ch = t + cc * 128;          // 0..511
                    int key = ch >> 3, db = ch & 7;
                    tv[cc] = *(const v8bf*)&Vb[kvbase +
                        (size_t)(kb * 64 + key) * EE + db * 8];
                }
                #pragma unroll
                for (int cc = 0; cc < 4; ++cc) {
                    int ch = t + cc * 128;
                    int key = ch >> 3, db = ch & 7;
                    #pragma unroll
                    for (int e = 0; e < 8; ++e)
                        Vtt[(db * 8 + e) * 64 + key] = tv[cc][e];
                }
            }
            if (wave == 0) __builtin_amdgcn_s_wait_tensorcnt(0);
            __syncthreads();

            // scores: 16q x 64k in 4 C tiles
            v8f st[4];
            #pragma unroll
            for (int kg = 0; kg < 4; ++kg) {
                v8f acc = {};
                #pragma unroll
                for (int kt = 0; kt < 2; ++kt) {
                    v16bf bf;
                    int key = kg * 16 + lq;
                    #pragma unroll
                    for (int e = 0; e < 16; ++e)
                        bf[e] = Kt[key * 64 + 32 * kt + 16 * half + e];
                    acc = __builtin_amdgcn_wmma_f32_16x16x32_bf16(
                        false, aq[kt], false, bf, (short)0, acc, false, false);
                }
                #pragma unroll
                for (int r = 0; r < 8; ++r) acc[r] *= wij;
                st[kg] = acc;
            }

            // online softmax (row m = r + 8*half; n spread over 16 lanes)
            float mnew[8], scl[8];
            #pragma unroll
            for (int r = 0; r < 8; ++r) {
                float mx = fmaxf(fmaxf(st[0][r], st[1][r]), fmaxf(st[2][r], st[3][r]));
                #pragma unroll
                for (int msk = 1; msk < 16; msk <<= 1)
                    mx = fmaxf(mx, __shfl_xor(mx, msk, 32));
                mnew[r] = fmaxf(mrun[r], mx);
                scl[r]  = __expf(mrun[r] - mnew[r]);
            }
            float rsum[8] = {};
            #pragma unroll
            for (int kg = 0; kg < 4; ++kg)
                #pragma unroll
                for (int r = 0; r < 8; ++r) {
                    float p = __expf(st[kg][r] - mnew[r]);
                    st[kg][r] = p;
                    rsum[r] += p;
                }
            #pragma unroll
            for (int r = 0; r < 8; ++r) {
                float s = rsum[r];
                #pragma unroll
                for (int msk = 1; msk < 16; msk <<= 1) s += __shfl_xor(s, msk, 32);
                lrun[r] = lrun[r] * scl[r] + s;
                mrun[r] = mnew[r];
            }
            #pragma unroll
            for (int nt = 0; nt < 4; ++nt)
                #pragma unroll
                for (int r = 0; r < 8; ++r) oj[nt][r] *= scl[r];

            // P (C layout) -> wave-private LDS -> A layout
            #pragma unroll
            for (int kg = 0; kg < 4; ++kg)
                #pragma unroll
                for (int r = 0; r < 8; ++r)
                    Pt[(wave * 16 + r + 8 * half) * 64 + kg * 16 + lq] = (__bf16)st[kg][r];
            asm volatile("s_wait_dscnt 0" ::: "memory");

            // O += P x V   (B-frag from transposed V: contiguous per lane)
            #pragma unroll
            for (int kt = 0; kt < 2; ++kt) {
                v16bf ap;
                #pragma unroll
                for (int e = 0; e < 16; ++e)
                    ap[e] = Pt[(wave * 16 + lq) * 64 + a_frag_k(e, half) + 32 * kt];
                #pragma unroll
                for (int nt = 0; nt < 4; ++nt) {
                    v16bf bv;
                    int d = nt * 16 + lq;
                    #pragma unroll
                    for (int e = 0; e < 16; ++e)
                        bv[e] = Vtt[d * 64 + 32 * kt + 16 * half + e];
                    oj[nt] = __builtin_amdgcn_wmma_f32_16x16x32_bf16(
                        false, ap, false, bv, (short)0, oj[nt], false, false);
                }
            }
        }
        #pragma unroll
        for (int nt = 0; nt < 4; ++nt)
            #pragma unroll
            for (int r = 0; r < 8; ++r)
                o[nt][r] += oj[nt][r] / (lrun[r] * (float)NSTR);
    }

    // write: mh2[i,b, t=h*64+d, e=q]  (reference's S==E transpose quirk)
    #pragma unroll
    for (int nt = 0; nt < 4; ++nt)
        #pragma unroll
        for (int r = 0; r < 8; ++r) {
            int d = nt * 16 + lq;
            int q = q0 + wave * 16 + r + 8 * half;
            size_t oidx = (((size_t)i * BB + b) * SS + (size_t)(h * HD + d)) * EE + q;
            mh2[oidx] = (__bf16)o[nt][r];
        }
}

// ---------------------------------------------------------------------------
__global__ __launch_bounds__(256) void add_ln(
    const float* __restrict__ A, const float* __restrict__ Bv,
    const float* __restrict__ g, const float* __restrict__ beta,
    float* __restrict__ outF, __bf16* __restrict__ outB, int rowsPerStream)
{
    int row = blockIdx.x;
    int ns  = row / rowsPerStream;
    size_t base = (size_t)row * EE;
    __shared__ float red[256];
    int t = threadIdx.x;

    float v0 = A[base + t]       + Bv[base + t];
    float v1 = A[base + t + 256] + Bv[base + t + 256];
    red[t] = v0 + v1;
    __syncthreads();
    for (int s = 128; s > 0; s >>= 1) { if (t < s) red[t] += red[t + s]; __syncthreads(); }
    float mean = red[0] * (1.f / EE);
    __syncthreads();
    float d0 = v0 - mean, d1 = v1 - mean;
    red[t] = d0 * d0 + d1 * d1;
    __syncthreads();
    for (int s = 128; s > 0; s >>= 1) { if (t < s) red[t] += red[t + s]; __syncthreads(); }
    float rstd = rsqrtf(red[0] * (1.f / EE) + 1e-5f);

    float y0 = d0 * rstd * g[ns * EE + t]       + beta[ns * EE + t];
    float y1 = d1 * rstd * g[ns * EE + t + 256] + beta[ns * EE + t + 256];
    if (outF) { outF[base + t] = y0; outF[base + t + 256] = y1; }
    if (outB) { outB[base + t] = (__bf16)y0; outB[base + t + 256] = (__bf16)y1; }
}

// ---------------------------------------------------------------------------
extern "C" void kernel_launch(void* const* d_in, const int* in_sizes, int n_in,
                              void* d_out, int out_size, void* d_ws, size_t ws_size,
                              hipStream_t stream) {
    (void)in_sizes; (void)n_in; (void)out_size; (void)ws_size;
    const float* x0 = (const float*)d_in[0];
    const float* x1 = (const float*)d_in[1];
    const float* x2 = (const float*)d_in[2];
    const float* x3 = (const float*)d_in[3];
    const float* Wq = (const float*)d_in[4];
    const float* Wk = (const float*)d_in[5];
    const float* Wv = (const float*)d_in[6];
    const float* Wo = (const float*)d_in[7];
    const float* bo = (const float*)d_in[8];
    const float* ln1g = (const float*)d_in[9];
    const float* ln1b = (const float*)d_in[10];
    const float* ln2g = (const float*)d_in[11];
    const float* ln2b = (const float*)d_in[12];
    const float* W1  = (const float*)d_in[13];
    const float* bf1 = (const float*)d_in[14];
    const float* W2  = (const float*)d_in[15];
    const float* bf2 = (const float*)d_in[16];
    const float* inter = (const float*)d_in[17];

    const int perStream = BB * SS * EE;            // 1,048,576
    const long total4   = (long)NSTR * perStream;  // 4,194,304
    const long wEE      = (long)NSTR * EE * EE;
    const long wE4E     = (long)NSTR * EE * 4 * EE;
    const long hTot     = (long)NSTR * BB * SS * 4 * EE;

    char* w = (char*)d_ws;
    auto carve = [&](size_t bytes) { char* p = w; w += (bytes + 255) & ~(size_t)255; return p; };
    float*  xs   = (float*) carve(total4 * 4);
    __bf16* xb   = (__bf16*)carve(total4 * 2);
    __bf16* qb   = (__bf16*)carve(total4 * 2);
    __bf16* kb   = (__bf16*)carve(total4 * 2);
    __bf16* vb   = (__bf16*)carve(total4 * 2);
    __bf16* Wqb  = (__bf16*)carve(wEE * 2);
    __bf16* Wkb  = (__bf16*)carve(wEE * 2);
    __bf16* Wvb  = (__bf16*)carve(wEE * 2);
    __bf16* Wob  = (__bf16*)carve(wEE * 2);
    __bf16* W1b  = (__bf16*)carve(wE4E * 2);
    __bf16* W2b  = (__bf16*)carve(wE4E * 2);
    __bf16* mh2b = (__bf16*)carve(total4 * 2);
    float*  proj = (float*) carve(total4 * 4);
    float*  r1   = (float*) carve(total4 * 4);
    __bf16* r1b  = (__bf16*)carve(total4 * 2);
    __bf16* hb   = (__bf16*)carve(hTot * 2);
    float*  ffn  = proj;   // proj dead after LN1; reuse for FFN output

    stack_cvt<<<dim3((perStream + 255) / 256, NSTR), 256, 0, stream>>>(
        x0, x1, x2, x3, xs, xb, perStream);
    cvt_bf16<<<(wEE  + 255) / 256, 256, 0, stream>>>(Wq, Wqb, wEE);
    cvt_bf16<<<(wEE  + 255) / 256, 256, 0, stream>>>(Wk, Wkb, wEE);
    cvt_bf16<<<(wEE  + 255) / 256, 256, 0, stream>>>(Wv, Wvb, wEE);
    cvt_bf16<<<(wEE  + 255) / 256, 256, 0, stream>>>(Wo, Wob, wEE);
    cvt_bf16<<<(wE4E + 255) / 256, 256, 0, stream>>>(W1, W1b, wE4E);
    cvt_bf16<<<(wE4E + 255) / 256, 256, 0, stream>>>(W2, W2b, wE4E);

    const int M = BB * SS;   // 2048
    gemm_bf16<<<dim3(EE / 64, M / 64, NSTR), 128, 0, stream>>>(
        xb, Wqb, nullptr, nullptr, qb, M, EE, EE, 0);
    gemm_bf16<<<dim3(EE / 64, M / 64, NSTR), 128, 0, stream>>>(
        xb, Wkb, nullptr, nullptr, kb, M, EE, EE, 0);
    gemm_bf16<<<dim3(EE / 64, M / 64, NSTR), 128, 0, stream>>>(
        xb, Wvb, nullptr, nullptr, vb, M, EE, EE, 0);

    attn_kernel<<<dim3(SS / 64, BB * HH, NSTR), 128, 0, stream>>>(
        qb, kb, vb, inter, mh2b);

    gemm_bf16<<<dim3(EE / 64, M / 64, NSTR), 128, 0, stream>>>(
        mh2b, Wob, bo, proj, nullptr, M, EE, EE, 0);

    add_ln<<<NSTR * BB * SS, 256, 0, stream>>>(xs, proj, ln1g, ln1b, r1, r1b, BB * SS);

    gemm_bf16<<<dim3(4 * EE / 64, M / 64, NSTR), 128, 0, stream>>>(
        r1b, W1b, bf1, nullptr, hb, M, 4 * EE, EE, 1);
    gemm_bf16<<<dim3(EE / 64, M / 64, NSTR), 128, 0, stream>>>(
        hb, W2b, bf2, ffn, nullptr, M, EE, 4 * EE, 0);

    add_ln<<<NSTR * BB * SS, 256, 0, stream>>>(
        r1, ffn, ln2g, ln2b, (float*)d_out, nullptr, BB * SS);
}
